// MambaTemporalEncoder_18047452578668
// MI455X (gfx1250) — matrix operand order repaired and checked by reading, not compiled, not run on hardware
//
#include <hip/hip_runtime.h>
#include <hip/hip_bf16.h>
#include <stdint.h>

// ---------------------------------------------------------------- constants
#define DM    256      // D_MODEL
#define DI    512      // D_INNER
#define DXZ   1024     // 2*D_INNER
#define DTR   16       // DT_RANK
#define DST   16       // D_STATE
#define DDBL  48       // DT_RANK + 2*D_STATE
#define TT    128      // T
#define BN    128      // B*N sequences
#define NROWS 16384    // BN*T
#define DCONV 4

typedef __attribute__((ext_vector_type(16))) _Float16 v16h;
typedef __attribute__((ext_vector_type(8)))  _Float16 v8h;
typedef __attribute__((ext_vector_type(8)))  float    v8f;

// ---------------------------------------------------------------- helpers
__device__ __forceinline__ float silu_f(float x) {
    return x / (1.0f + __expf(-x));
}

__device__ __forceinline__ v16h load_a_frag(const _Float16* __restrict__ p) {
    // A fragment: one contiguous 32B chunk {k..k+15}
    v8h lo = *(const v8h*)(p);
    v8h hi = *(const v8h*)(p + 8);
    return __builtin_shufflevector(lo, hi, 0, 1, 2, 3, 4, 5, 6, 7,
                                           8, 9, 10, 11, 12, 13, 14, 15);
}

// ================================================================ kernels

// Gather x (B,T,N,DM) into xr row-major rows r=(b*N+n)*T+t, converted to f16.
__global__ __launch_bounds__(256)
void gather_x_f16(const float* __restrict__ x, _Float16* __restrict__ a16) {
    size_t i = (size_t)blockIdx.x * blockDim.x + threadIdx.x;   // < NROWS*DM
    if (i >= (size_t)NROWS * DM) return;
    int    c  = (int)(i & (DM - 1));
    size_t r  = i >> 8;                 // row: (b*64+n)*128 + t
    int    t  = (int)(r & (TT - 1));
    size_t bn = r >> 7;
    int    n  = (int)(bn & 63);
    int    b  = (int)(bn >> 6);
    size_t src = ((((size_t)b * TT + t) * 64) + n) * DM + c;
    a16[i] = (_Float16)x[src];
}

// Generic f32 -> f16 copy (for weight matrices).
__global__ __launch_bounds__(256)
void cvt_f32_to_f16(const float* __restrict__ src, _Float16* __restrict__ dst, int n) {
    int i = blockIdx.x * blockDim.x + threadIdx.x;
    if (i < n) dst[i] = (_Float16)src[i];
}

// C[M,N] = A[M,K] * W[N,K]^T, K must be a multiple of 64.
// One wave computes a 16 x (16*NT) strip: one A fragment is reused against NT
// B fragments per k-step (register blocking along N).  The k-loop is manually
// unrolled x2 with explicit ping-pong buffers: each buffer's loads are issued
// a half-iteration ahead of its WMMA burst, and there is no buffer-rotation
// code for the compiler to lower into v_mov copies.
template <int NT>
__global__ __launch_bounds__(32)
void gemm_f16_wmma(const _Float16* __restrict__ A, const _Float16* __restrict__ W,
                   float* __restrict__ C, int M, int N, int K) {
    int ntn  = N / (16 * NT);        // strip columns
    int tile = blockIdx.x;
    int tm   = tile / ntn;
    int tj   = tile - tm * ntn;
    int tn0  = tj * NT;              // first n-tile of this strip
    int lane = threadIdx.x;
    int mn   = lane & 15;            // row-of-A / row-of-W handled by this lane
    int half = lane >> 4;            // lane group 0/1

    const _Float16* arow = A + (size_t)(tm * 16 + mn) * K + half * 16;
    const _Float16* brow[NT];
#pragma unroll
    for (int j = 0; j < NT; ++j)
        brow[j] = W + (size_t)((tn0 + j) * 16 + mn) * K + half * 16;

    v8f acc[NT];
#pragma unroll
    for (int j = 0; j < NT; ++j) acc[j] = (v8f){};

    // prologue: buffer 0 <- k = 0
    v16h a0 = load_a_frag(arow);
    v16h b0[NT];
#pragma unroll
    for (int j = 0; j < NT; ++j) b0[j] = *(const v16h*)(brow[j]);

    v16h a1;
    v16h b1[NT];

    int k = 0;
    for (; k < K - 64; k += 64) {
        // prime GL2 a pair ahead (global_prefetch_b8)
        __builtin_prefetch(arow + k + 64, 0, 3);
        __builtin_prefetch(brow[0] + k + 64, 0, 3);
        // buffer 1 <- k+32, compute on buffer 0 (k)
        a1 = load_a_frag(arow + k + 32);
#pragma unroll
        for (int j = 0; j < NT; ++j) b1[j] = *(const v16h*)(brow[j] + k + 32);
#pragma unroll
        for (int j = 0; j < NT; ++j)
            acc[j] = __builtin_amdgcn_wmma_f32_16x16x32_f16(
                false, a0, false, b0[j], (short)0, acc[j], false, false);
        // buffer 0 <- k+64, compute on buffer 1 (k+32)
        a0 = load_a_frag(arow + k + 64);
#pragma unroll
        for (int j = 0; j < NT; ++j) b0[j] = *(const v16h*)(brow[j] + k + 64);
#pragma unroll
        for (int j = 0; j < NT; ++j)
            acc[j] = __builtin_amdgcn_wmma_f32_16x16x32_f16(
                false, a1, false, b1[j], (short)0, acc[j], false, false);
    }
    // tail pair: k = K-64 (buffer 0 already holds it), K-32 into buffer 1
    a1 = load_a_frag(arow + k + 32);
#pragma unroll
    for (int j = 0; j < NT; ++j) b1[j] = *(const v16h*)(brow[j] + k + 32);
#pragma unroll
    for (int j = 0; j < NT; ++j)
        acc[j] = __builtin_amdgcn_wmma_f32_16x16x32_f16(
            false, a0, false, b0[j], (short)0, acc[j], false, false);
#pragma unroll
    for (int j = 0; j < NT; ++j)
        acc[j] = __builtin_amdgcn_wmma_f32_16x16x32_f16(
            false, a1, false, b1[j], (short)0, acc[j], false, false);

#pragma unroll
    for (int j = 0; j < NT; ++j) {
        float* crow = C + (size_t)(tm * 16 + half * 8) * N + (tn0 + j) * 16 + mn;
#pragma unroll
        for (int v = 0; v < 8; ++v) crow[(size_t)v * N] = acc[j][v];
    }
}

// Depthwise causal conv (k=4) + bias + SiLU.  xp = xz[:, 0:DI] (row stride DXZ).
// Writes u (f32) and u16 (f16) at [row*DI + d].
__global__ __launch_bounds__(256)
void conv_silu(const float* __restrict__ xz, const float* __restrict__ Wc,
               const float* __restrict__ bc, float* __restrict__ u,
               _Float16* __restrict__ u16) {
    size_t i = (size_t)blockIdx.x * blockDim.x + threadIdx.x;   // < NROWS*DI
    if (i >= (size_t)NROWS * DI) return;
    int    d   = (int)(i & (DI - 1));
    size_t row = i >> 9;
    int    t   = (int)(row & (TT - 1));
    float acc = bc[d];
#pragma unroll
    for (int j = 0; j < DCONV; ++j) {
        int tt = t - (DCONV - 1) + j;
        if (tt >= 0)
            acc += Wc[d * DCONV + j] * xz[(row - (DCONV - 1) + j) * (size_t)DXZ + d];
    }
    float s = silu_f(acc);
    u[i]   = s;
    u16[i] = (_Float16)s;
}

// Fused dt-projection + softplus + selective scan + D skip + z gate.
// One block per sequence (BN), one thread per channel d (DI). State in regs.
__global__ __launch_bounds__(512)
void scan_fused(const float* __restrict__ xdbl, const float* __restrict__ u,
                const float* __restrict__ xz, const float* __restrict__ A_log,
                const float* __restrict__ W_dt, const float* __restrict__ b_dt,
                const float* __restrict__ D_param, _Float16* __restrict__ yg16) {
    __shared__ float sdbl[DDBL];
    int b = blockIdx.x;      // sequence 0..BN-1
    int d = threadIdx.x;     // channel 0..DI-1

    float Arow[DST], Wdt[DTR], h[DST];
#pragma unroll
    for (int s = 0; s < DST; ++s) { Arow[s] = -__expf(A_log[d * DST + s]); h[s] = 0.0f; }
#pragma unroll
    for (int r = 0; r < DTR; ++r) Wdt[r] = W_dt[d * DTR + r];
    float bdt = b_dt[d];
    float Dd  = D_param[d];

    for (int t = 0; t < TT; ++t) {
        size_t row = (size_t)b * TT + t;
        __syncthreads();
        if (d < DDBL) sdbl[d] = xdbl[row * DDBL + d];
        __syncthreads();

        // dt = softplus(dt_raw @ W_dt^T + b_dt)   (K = DT_RANK = 16)
        float dtr = bdt;
#pragma unroll
        for (int r = 0; r < DTR; ++r) dtr += Wdt[r] * sdbl[r];
        float dtv = (dtr > 20.0f) ? dtr : __logf(1.0f + __expf(dtr));

        float ut = u[row * DI + d];
        float y  = 0.0f;
#pragma unroll
        for (int s = 0; s < DST; ++s) {
            float dA = __expf(dtv * Arow[s]);
            h[s] = h[s] * dA + (dtv * sdbl[DTR + s]) * ut;   // B_t
            y   += h[s] * sdbl[DTR + DST + s];               // C_t
        }
        y += ut * Dd;                         // D skip
        float zv = xz[row * DXZ + DI + d];    // z half of xz
        y *= silu_f(zv);                      // gate
        yg16[row * DI + d] = (_Float16)y;
    }
}

// LayerNorm over DM + gamma/beta, writing to output layout (B,T,N,DM).
__global__ __launch_bounds__(256)
void layernorm_permute(const float* __restrict__ outlin, const float* __restrict__ gamma,
                       const float* __restrict__ beta, float* __restrict__ out) {
    __shared__ float sbuf[DM];
    int row = blockIdx.x;    // 0..NROWS-1 : (b*64+n)*128 + t
    int c   = threadIdx.x;
    float v = outlin[(size_t)row * DM + c];

    sbuf[c] = v; __syncthreads();
    for (int s = DM / 2; s > 0; s >>= 1) {
        if (c < s) sbuf[c] += sbuf[c + s];
        __syncthreads();
    }
    float mu = sbuf[0] / (float)DM;
    __syncthreads();
    float dv = v - mu;
    sbuf[c] = dv * dv; __syncthreads();
    for (int s = DM / 2; s > 0; s >>= 1) {
        if (c < s) sbuf[c] += sbuf[c + s];
        __syncthreads();
    }
    float var  = sbuf[0] / (float)DM;
    float rstd = rsqrtf(var + 1e-5f);

    int t  = row & (TT - 1);
    int bn = row >> 7;
    int n  = bn & 63;
    int b  = bn >> 6;
    size_t o = ((((size_t)b * TT + t) * 64) + n) * DM + c;
    out[o] = dv * rstd * gamma[c] + beta[c];
}

// ================================================================ host
static inline size_t align256(size_t x) { return (x + 255) & ~(size_t)255; }

extern "C" void kernel_launch(void* const* d_in, const int* in_sizes, int n_in,
                              void* d_out, int out_size, void* d_ws, size_t ws_size,
                              hipStream_t stream) {
    (void)in_sizes; (void)n_in; (void)out_size; (void)ws_size;
    const float* x       = (const float*)d_in[0];
    const float* W_in    = (const float*)d_in[1];
    const float* W_conv  = (const float*)d_in[2];
    const float* b_conv  = (const float*)d_in[3];
    const float* W_xproj = (const float*)d_in[4];
    const float* W_dt    = (const float*)d_in[5];
    const float* b_dt    = (const float*)d_in[6];
    const float* A_log   = (const float*)d_in[7];
    const float* D_param = (const float*)d_in[8];
    const float* W_out   = (const float*)d_in[9];
    const float* gamma   = (const float*)d_in[10];
    const float* beta    = (const float*)d_in[11];
    float* out = (float*)d_out;

    // -------- workspace carve-up
    uint8_t* ws = (uint8_t*)d_ws;
    size_t off = 0;
    _Float16* a16     = (_Float16*)(ws + off); off = align256(off + (size_t)NROWS * DM * 2);
    _Float16* win16   = (_Float16*)(ws + off); off = align256(off + (size_t)DXZ * DM * 2);
    _Float16* wxp16   = (_Float16*)(ws + off); off = align256(off + (size_t)DDBL * DI * 2);
    _Float16* wout16  = (_Float16*)(ws + off); off = align256(off + (size_t)DM * DI * 2);
    float*    xz      = (float*)   (ws + off); off = align256(off + (size_t)NROWS * DXZ * 4);
    float*    u       = (float*)   (ws + off); off = align256(off + (size_t)NROWS * DI * 4);
    _Float16* u16     = (_Float16*)(ws + off); off = align256(off + (size_t)NROWS * DI * 2);
    float*    xdbl    = (float*)   (ws + off); off = align256(off + (size_t)NROWS * DDBL * 4);
    _Float16* yg16    = (_Float16*)(ws + off); off = align256(off + (size_t)NROWS * DI * 2);
    float*    outlin  = (float*)   (ws + off); off = align256(off + (size_t)NROWS * DM * 4);

    // 1) gather + f16 convert activations / weights
    gather_x_f16<<<(NROWS * DM) / 256, 256, 0, stream>>>(x, a16);
    cvt_f32_to_f16<<<(DXZ * DM + 255) / 256, 256, 0, stream>>>(W_in, win16, DXZ * DM);
    cvt_f32_to_f16<<<(DDBL * DI + 255) / 256, 256, 0, stream>>>(W_xproj, wxp16, DDBL * DI);
    cvt_f32_to_f16<<<(DM * DI + 255) / 256, 256, 0, stream>>>(W_out, wout16, DM * DI);

    // 2) xz = xr @ W_in^T           (16384x256 @ 256x1024), 16x64 strips
    gemm_f16_wmma<4><<<(NROWS / 16) * (DXZ / 64), 32, 0, stream>>>(a16, win16, xz,
                                                                   NROWS, DXZ, DM);
    // 3) depthwise causal conv + SiLU
    conv_silu<<<(NROWS * DI) / 256, 256, 0, stream>>>(xz, W_conv, b_conv, u, u16);

    // 4) x_dbl = u @ W_xproj^T      (16384x512 @ 512x48), 16x48 strips
    gemm_f16_wmma<3><<<(NROWS / 16) * (DDBL / 48), 32, 0, stream>>>(u16, wxp16, xdbl,
                                                                    NROWS, DDBL, DI);
    // 5) fused dt + selective scan + D skip + z gate
    scan_fused<<<BN, DI, 0, stream>>>(xdbl, u, xz, A_log, W_dt, b_dt, D_param, yg16);

    // 6) out_lin = y @ W_out^T      (16384x512 @ 512x256), 16x64 strips
    gemm_f16_wmma<4><<<(NROWS / 16) * (DM / 64), 32, 0, stream>>>(yg16, wout16, outlin,
                                                                  NROWS, DM, DI);
    // 7) LayerNorm + permute to (B,T,N,DM)
    layernorm_permute<<<NROWS, DM, 0, stream>>>(outlin, gamma, beta, out);
}
